// Actor_77197742178495
// MI455X (gfx1250) — matrix operand order repaired and checked by reading, not compiled, not run on hardware
//
#include <hip/hip_runtime.h>

#define BS 8
#define T 128
#define HW 160
#define HID 128
#define FEAT 512
#define ACT 18
#define NFRAMES (BS * T) // 1024

typedef __attribute__((ext_vector_type(16))) __bf16 v16bf;
typedef __attribute__((ext_vector_type(8)))  float  v8f;
typedef __attribute__((ext_vector_type(8)))  unsigned short u16x8;
typedef __attribute__((ext_vector_type(4)))  unsigned short u16x4;

union BFx16 { v16bf v; u16x8 h8[2]; u16x4 h4[4]; unsigned short u[16]; };
union Fx8   { v8f   v; float f[8]; };

__device__ __forceinline__ unsigned short f2bf(float f) {
    unsigned u = __builtin_bit_cast(unsigned, f);
    return (unsigned short)((u + 0x7FFFu + ((u >> 16) & 1u)) >> 16);
}

__device__ __forceinline__ v8f wmma_bf16(v16bf a, v16bf b, v8f c) {
    return __builtin_amdgcn_wmma_f32_16x16x32_bf16(false, a, false, b, (short)0, c, false, false);
}

__device__ __forceinline__ Fx8 zero8() {
    Fx8 z;
#pragma unroll
    for (int g = 0; g < 8; ++g) z.f[g] = 0.0f;
    return z;
}

// Async global->LDS copy of 16 bytes per active lane (ASYNCcnt-tracked).
// LDS byte offset = low 32 bits of the generic shared pointer (LDS aperture).
__device__ __forceinline__ void async_copy_b128(const unsigned short* gsrc,
                                                unsigned short* ldst) {
    unsigned ldsoff = (unsigned)(size_t)ldst;
    asm volatile("global_load_async_to_lds_b128 %0, %1, off"
                 :: "v"(ldsoff), "v"(gsrc) : "memory");
}
__device__ __forceinline__ void async_wait0() {
    asm volatile("s_wait_asynccnt 0x0" ::: "memory");
}

// WMMA 16-bit tile layouts (wave32):
//  A (16x32 MxK): lane m = lane&15, ksel = lane>>4; two contiguous 8-elem runs:
//     k = kb + ksel*8 + [0..7]  and  k = kb + 16 + ksel*8 + [0..7]
//  B (32x16 KxN): lane n = lane&15, ksel = lane>>4; two contiguous 8-elem runs:
//     k = kb + ksel*8 + [0..7]  and  k = kb + 16 + ksel*8 + [0..7]
//  D (16x16 f32): lane n = lane&15; vgpr g -> m = g + (lane>>4)*8

__device__ __forceinline__ v16bf load_b16(const unsigned short* p) { // p[0..7], p[16..23]
    BFx16 b;
    b.h8[0] = *(const u16x8*)(p);
    b.h8[1] = *(const u16x8*)(p + 16);
    return b.v;
}
__device__ __forceinline__ v16bf load_a_row(const unsigned short* row, int kb, int ksel) {
    BFx16 a;
    a.h8[0] = *(const u16x8*)(row + kb + (ksel << 3));
    a.h8[1] = *(const u16x8*)(row + kb + 16 + (ksel << 3));
    return a.v;
}

// ---------------------------------------------------------------------------
// fp32 -> bf16 conversion with zero padding (weight pre-conversion)
__global__ void cvt_bf16_kernel(const float* __restrict__ src,
                                unsigned short* __restrict__ dst,
                                int nSrc, int nDst) {
    int i = blockIdx.x * 256 + threadIdx.x;
    if (i < nDst) dst[i] = (i < nSrc) ? f2bf(src[i]) : (unsigned short)0;
}

// ---------------------------------------------------------------------------
// conv1: x[1024,160,160] -> feat1[1024,32,400]; 16x16 k, stride 8, pad 4
// GEMM per frame: M=400 (25 tiles), N=32 (2 tiles), K=256
// Zero-padded fp32->bf16 slab staged in LDS: 24 rows x 168 cols
__global__ void conv1_kernel(const float* __restrict__ x,
                             const unsigned short* __restrict__ w, // bf16 [32,256]
                             const float* __restrict__ bias,
                             unsigned short* __restrict__ feat1) {
    __shared__ __align__(16) unsigned short tile[24 * 168];
    const int frame = blockIdx.x;
    const int mt    = blockIdx.y;           // 0..24
    const int tid   = threadIdx.x;          // 64 threads
    const int wv    = tid >> 5;             // N tile 0..1
    const int lane  = tid & 31;
    const int mrow  = lane & 15;
    const int ksel  = lane >> 4;
    const int pos   = mt * 16 + mrow;       // 0..399
    const int oy    = pos / 20, ox = pos % 20;
    const int n     = wv * 16 + mrow;
    const int oyB   = (mt * 16) / 20;       // base output row of this tile
    const float* xf = x + (size_t)frame * (HW * HW);

    // stage zero-padded slab: rows iy = oyB*8-4 .. +23, cols ix = -4 .. 163
    for (int idx = tid; idx < 24 * 168; idx += 64) {
        int ly = idx / 168, lx = idx % 168;
        int iy = oyB * 8 - 4 + ly;
        int ix = lx - 4;
        int iyc = iy < 0 ? 0 : (iy > HW - 1 ? HW - 1 : iy);
        int ixc = ix < 0 ? 0 : (ix > HW - 1 ? HW - 1 : ix);
        float v = xf[iyc * HW + ixc];                       // unconditional load
        bool ok = ((unsigned)iy < HW) && ((unsigned)ix < HW);
        tile[idx] = ok ? f2bf(v) : (unsigned short)0;
    }
    __syncthreads();

    const int dy8 = (oy - oyB) * 8;
    const unsigned short* wrow = w + n * 256;
    Fx8 acc = zero8();
#pragma unroll
    for (int kb = 0; kb < 256; kb += 32) {
        int off = (dy8 + (kb >> 4)) * 168 + ox * 8 + (ksel << 3);
        BFx16 a;
        a.h8[0] = *(const u16x8*)(tile + off);
        a.h8[1] = *(const u16x8*)(tile + off + 168);
        v16bf b = load_b16(wrow + kb + (ksel << 3));
        acc.v = wmma_bf16(a.v, b, acc.v);
    }
    const float bv = bias[n];
#pragma unroll
    for (int g = 0; g < 8; ++g) {
        int   p = mt * 16 + g + (ksel << 3);
        float v = acc.f[g] + bv;
        v = v > 0.0f ? v : 0.0f;
        feat1[(size_t)frame * 12800 + (size_t)n * 400 + p] = f2bf(v);
    }
}

// ---------------------------------------------------------------------------
// conv2: feat1[1024,32,20,20] -> feat2[1024,64,25]; 8x8 k, stride 4, pad 2
// Raw frame (25600 B) async-copied global->LDS, then zero-padded im2col
// expansion done LDS->LDS. GEMM: M=25 (2 tiles), N=64 (4 tiles), K=2048
__global__ void conv2_kernel(const unsigned short* __restrict__ feat1,
                             const unsigned short* __restrict__ w, // bf16 [64,2048]
                             const float* __restrict__ bias,
                             unsigned short* __restrict__ feat2) {
    __shared__ __align__(16) unsigned short raw[32 * 400];  // [c][20][20]
    __shared__ __align__(16) unsigned short tile[32 * 576]; // [c][24][24]
    const int frame = blockIdx.x;
    const int tid   = threadIdx.x;            // 256 threads
    const int wv    = tid >> 5;               // 0..7
    const int mt    = wv >> 2;                // 0..1
    const int nt    = wv & 3;                 // 0..3
    const int lane  = tid & 31;
    const int mrow  = lane & 15;
    const int ksel  = lane >> 4;
    int pos = mt * 16 + mrow; if (pos > 24) pos = 24;
    const int oy = pos / 5, ox = pos % 5;
    const int n  = nt * 16 + mrow;
    const unsigned short* f1 = feat1 + (size_t)frame * 12800;

    // async bulk copy: 1600 x 16B
    for (int idx = tid; idx < 1600; idx += 256)
        async_copy_b128(f1 + idx * 8, raw + idx * 8);
    async_wait0();
    __syncthreads();

    // padded expansion, LDS -> LDS
    for (int idx = tid; idx < 32 * 576; idx += 256) {
        int c = idx / 576, r = idx % 576;
        int ly = r / 24, lx = r % 24;
        int iy = ly - 2, ix = lx - 2;
        int iyc = iy < 0 ? 0 : (iy > 19 ? 19 : iy);
        int ixc = ix < 0 ? 0 : (ix > 19 ? 19 : ix);
        unsigned short v = raw[c * 400 + iyc * 20 + ixc];
        bool ok = ((unsigned)iy < 20u) && ((unsigned)ix < 20u);
        tile[idx] = ok ? v : (unsigned short)0;
    }
    __syncthreads();

    const unsigned short* wrow = w + (size_t)n * 2048;
    Fx8 acc = zero8();
#pragma unroll 4
    for (int kb = 0; kb < 2048; kb += 32) {
        int cb = kb >> 6;                       // channel (constant over k-tile)
        int r0 = (kb & 63) + (ksel << 3);       // chunk0: ky = r0>>3, kx = 0..7
        int r1 = r0 + 16;                       // chunk1
        int base0 = cb * 576 + (oy * 4 + (r0 >> 3)) * 24 + ox * 4;
        int base1 = cb * 576 + (oy * 4 + (r1 >> 3)) * 24 + ox * 4;
        BFx16 a;
        a.h4[0] = *(const u16x4*)(tile + base0);
        a.h4[1] = *(const u16x4*)(tile + base0 + 4);
        a.h4[2] = *(const u16x4*)(tile + base1);
        a.h4[3] = *(const u16x4*)(tile + base1 + 4);
        v16bf b = load_b16(wrow + kb + (ksel << 3));
        acc.v = wmma_bf16(a.v, b, acc.v);
    }
    const float bv = bias[n];
#pragma unroll
    for (int g = 0; g < 8; ++g) {
        int m = mt * 16 + g + (ksel << 3);
        if (m < 25) {
            float v = acc.f[g] + bv;
            v = v > 0.0f ? v : 0.0f;
            feat2[(size_t)frame * 1600 + n * 25 + m] = f2bf(v);
        }
    }
}

// ---------------------------------------------------------------------------
// conv3: feat2[1024,64,5,5] -> feat3[1024,512]; 3x3 k, stride 2, pad 0
// Whole input frame (3200 B) async-copied to LDS. M=4 (1 tile), N=128 (8), K=576
__global__ void conv3_kernel(const unsigned short* __restrict__ feat2,
                             const unsigned short* __restrict__ w, // bf16 [128,576]
                             const float* __restrict__ bias,
                             unsigned short* __restrict__ feat3) {
    __shared__ __align__(16) unsigned short tile[1600];
    const int frame = blockIdx.x;
    const int tid   = threadIdx.x;       // 256
    const int wv    = tid >> 5;          // N tile 0..7
    const int lane  = tid & 31;
    const int mrow  = lane & 15;
    const int ksel  = lane >> 4;
    int pos = mrow > 3 ? 3 : mrow;
    const int oy = pos >> 1, ox = pos & 1;
    const int n  = wv * 16 + mrow;

    const unsigned short* f2 = feat2 + (size_t)frame * 1600;
    if (tid < 200) async_copy_b128(f2 + tid * 8, tile + tid * 8); // 200 x 16B
    async_wait0();
    __syncthreads();

    const unsigned short* wrow = w + (size_t)n * 576;
    Fx8 acc = zero8();
#pragma unroll 6
    for (int kb = 0; kb < 576; kb += 32) {
        BFx16 a;
#pragma unroll
        for (int e = 0; e < 16; ++e) {          // k/9 split: gather from LDS
            int k  = kb + ((e >> 3) << 4) + (ksel << 3) + (e & 7);
            int c  = k / 9, r = k % 9;
            a.u[e] = tile[c * 25 + (oy * 2 + r / 3) * 5 + (ox * 2 + r % 3)];
        }
        v16bf b = load_b16(wrow + kb + (ksel << 3));
        acc.v = wmma_bf16(a.v, b, acc.v);
    }
    const float bv = bias[n];
#pragma unroll
    for (int g = 0; g < 8; ++g) {
        int ml = g + (ksel << 3);
        if (ml < 4) {
            float v = acc.f[g] + bv;
            v = v > 0.0f ? v : 0.0f;
            feat3[(size_t)frame * 512 + n * 4 + ml] = f2bf(v);
        }
    }
}

// ---------------------------------------------------------------------------
// fc1: relu(feat3[1024,512] @ fc1_w^T + b) -> fout bf16 [1024,512]
__global__ void fc1_kernel(const unsigned short* __restrict__ A,
                           const unsigned short* __restrict__ W, // bf16 [512,512]
                           const float* __restrict__ bias,
                           unsigned short* __restrict__ out) {
    const int mt   = blockIdx.x;
    const int wv   = threadIdx.x >> 5;
    const int nt   = blockIdx.y * 4 + wv;     // 0..31
    const int lane = threadIdx.x & 31;
    const int mrow = lane & 15;
    const int ksel = lane >> 4;
    const int n    = nt * 16 + mrow;
    const unsigned short* Ar = A + (size_t)(mt * 16 + mrow) * 512;
    const unsigned short* Wr = W + (size_t)n * 512;

    Fx8 acc = zero8();
#pragma unroll
    for (int kb = 0; kb < 512; kb += 32) {
        v16bf a = load_a_row(Ar, kb, ksel);
        v16bf b = load_b16(Wr + kb + (ksel << 3));
        acc.v = wmma_bf16(a, b, acc.v);
    }
    const float bv = bias[n];
#pragma unroll
    for (int g = 0; g < 8; ++g) {
        int   m = mt * 16 + g + (ksel << 3);
        float v = acc.f[g] + bv;
        v = v > 0.0f ? v : 0.0f;
        out[(size_t)m * 512 + n] = f2bf(v);
    }
}

// ---------------------------------------------------------------------------
// gru_x: gx[1024,384] = fout @ [Wz;Wr;Wh]^T + [bz;br;bh]  (fp32 out)
__global__ void grux_kernel(const unsigned short* __restrict__ A,
                            const unsigned short* __restrict__ Wcat, // bf16 [384,512]
                            const float* __restrict__ bz,
                            const float* __restrict__ br,
                            const float* __restrict__ bh,
                            float* __restrict__ gx) {
    const int mt   = blockIdx.x;
    const int wv   = threadIdx.x >> 5;
    const int nt   = blockIdx.y * 4 + wv;     // 0..23
    const int lane = threadIdx.x & 31;
    const int mrow = lane & 15;
    const int ksel = lane >> 4;
    const int n    = nt * 16 + mrow;          // 0..383
    const int nl   = n & 127;
    const float* bp = (nt < 8) ? bz : (nt < 16) ? br : bh;
    const unsigned short* Ar = A + (size_t)(mt * 16 + mrow) * 512;
    const unsigned short* Wr = Wcat + (size_t)n * 512;

    Fx8 acc = zero8();
#pragma unroll
    for (int kb = 0; kb < 512; kb += 32) {
        v16bf a = load_a_row(Ar, kb, ksel);
        v16bf b = load_b16(Wr + kb + (ksel << 3));
        acc.v = wmma_bf16(a, b, acc.v);
    }
    const float bv = bp[nl];
#pragma unroll
    for (int g = 0; g < 8; ++g) {
        int m = mt * 16 + g + (ksel << 3);
        gx[(size_t)m * 384 + n] = acc.f[g] + bv;
    }
}

// ---------------------------------------------------------------------------
// GRU scan: 1 block, 8 waves; h in LDS (bf16+f32), U matrices in VGPRs
__global__ void gru_scan_kernel(const float* __restrict__ gx,
                                const unsigned short* __restrict__ Ucat, // bf16 [3,128,128]
                                unsigned short* __restrict__ hiddenb,    // bf16 [8,128,128]
                                float* __restrict__ hlast) {             // f32 [8,128]
    __shared__ __align__(16) unsigned short hbf[16 * 128];
    __shared__ __align__(16) unsigned short rhbf[16 * 128];
    __shared__ float hf[8 * 128];

    const int tid  = threadIdx.x;
    const int wv   = tid >> 5;
    const int lane = tid & 31;
    const int mrow = lane & 15;
    const int ksel = lane >> 4;
    const int n    = wv * 16 + mrow;

    for (int i = tid; i < 16 * 128; i += 256) { hbf[i] = 0; rhbf[i] = 0; }
    for (int i = tid; i < 8 * 128;  i += 256) hf[i] = 0.0f;
    __syncthreads();

    // preload U B-tiles into registers: 3 x 4 x v16bf = 96 VGPRs
    const unsigned short* uz = Ucat + (size_t)n * 128;
    const unsigned short* ur = uz + 128 * 128;
    const unsigned short* uh = ur + 128 * 128;
    v16bf Bz[4], Br[4], Bh[4];
#pragma unroll
    for (int ki = 0; ki < 4; ++ki) {
        Bz[ki] = load_b16(uz + ki * 32 + (ksel << 3));
        Br[ki] = load_b16(ur + ki * 32 + (ksel << 3));
        Bh[ki] = load_b16(uh + ki * 32 + (ksel << 3));
    }

    for (int t = 0; t < T; ++t) {
        Fx8 accz = zero8(), accr = zero8();
#pragma unroll
        for (int ki = 0; ki < 4; ++ki) {
            v16bf a = load_a_row(hbf + mrow * 128, ki * 32, ksel);
            accz.v = wmma_bf16(a, Bz[ki], accz.v);
            accr.v = wmma_bf16(a, Br[ki], accr.v);
        }
        float zreg[8], hold[8];
#pragma unroll
        for (int g = 0; g < 8; ++g) {
            zreg[g] = 0.0f; hold[g] = 0.0f;
            int m = g + (ksel << 3);
            if (m < 8) {
                const float* gxr = gx + (size_t)(m * T + t) * 384;
                float z = 1.0f / (1.0f + __expf(-(accz.f[g] + gxr[n])));
                float r = 1.0f / (1.0f + __expf(-(accr.f[g] + gxr[128 + n])));
                float h = hf[m * 128 + n];
                zreg[g] = z; hold[g] = h;
                rhbf[m * 128 + n] = f2bf(r * h);
            }
        }
        __syncthreads();
        Fx8 acch = zero8();
#pragma unroll
        for (int ki = 0; ki < 4; ++ki) {
            v16bf a = load_a_row(rhbf + mrow * 128, ki * 32, ksel);
            acch.v = wmma_bf16(a, Bh[ki], acch.v);
        }
#pragma unroll
        for (int g = 0; g < 8; ++g) {
            int m = g + (ksel << 3);
            if (m < 8) {
                float hh = tanhf(acch.f[g] + gx[(size_t)(m * T + t) * 384 + 256 + n]);
                float hn = (1.0f - zreg[g]) * hold[g] + zreg[g] * hh;
                hf[m * 128 + n]  = hn;
                unsigned short hb = f2bf(hn);
                hbf[m * 128 + n] = hb;
                hiddenb[(size_t)(m * T + t) * 128 + n] = hb;
            }
        }
        __syncthreads();
    }
#pragma unroll
    for (int g = 0; g < 8; ++g) {
        int m = g + (ksel << 3);
        if (m < 8) hlast[m * 128 + n] = hf[m * 128 + n];
    }
}

// ---------------------------------------------------------------------------
// fc3: out[1024,18] = hidden[1024,128] @ fc3_w^T + b  (W padded to 32 rows)
__global__ void fc3_kernel(const unsigned short* __restrict__ hiddenb,
                           const unsigned short* __restrict__ W, // bf16 [32,128] padded
                           const float* __restrict__ bias,
                           float* __restrict__ out) {
    const int mt   = blockIdx.x;
    const int wv   = threadIdx.x >> 5;       // N tile 0..1
    const int lane = threadIdx.x & 31;
    const int mrow = lane & 15;
    const int ksel = lane >> 4;
    const int n    = wv * 16 + mrow;         // 0..31, valid < 18
    const unsigned short* Ar = hiddenb + (size_t)(mt * 16 + mrow) * 128;
    const unsigned short* Wr = W + (size_t)n * 128;

    Fx8 acc = zero8();
#pragma unroll
    for (int kb = 0; kb < 128; kb += 32) {
        v16bf a = load_a_row(Ar, kb, ksel);
        v16bf b = load_b16(Wr + kb + (ksel << 3));
        acc.v = wmma_bf16(a, b, acc.v);
    }
    if (n < ACT) {
        const float bv = bias[n];
#pragma unroll
        for (int g = 0; g < 8; ++g) {
            int m = mt * 16 + g + (ksel << 3);
            out[(size_t)m * ACT + n] = acc.f[g] + bv;
        }
    }
}

// ---------------------------------------------------------------------------
static inline void cvt(const float* s, unsigned short* d, int nS, int nD, hipStream_t st) {
    cvt_bf16_kernel<<<(nD + 255) / 256, 256, 0, st>>>(s, d, nS, nD);
}

extern "C" void kernel_launch(void* const* d_in, const int* in_sizes, int n_in,
                              void* d_out, int out_size, void* d_ws, size_t ws_size,
                              hipStream_t stream) {
    const float* x       = (const float*)d_in[0];
    const float* conv1_w = (const float*)d_in[1];
    const float* conv1_b = (const float*)d_in[2];
    const float* conv2_w = (const float*)d_in[3];
    const float* conv2_b = (const float*)d_in[4];
    const float* conv3_w = (const float*)d_in[5];
    const float* conv3_b = (const float*)d_in[6];
    const float* fc1_w   = (const float*)d_in[7];
    const float* fc1_b   = (const float*)d_in[8];
    const float* Wz = (const float*)d_in[9];   const float* bz = (const float*)d_in[10];
    const float* Uz = (const float*)d_in[11];
    const float* Wr = (const float*)d_in[12];  const float* br = (const float*)d_in[13];
    const float* Ur = (const float*)d_in[14];
    const float* Wh = (const float*)d_in[15];  const float* bh = (const float*)d_in[16];
    const float* Uh = (const float*)d_in[17];
    const float* fc3_w = (const float*)d_in[18];
    const float* fc3_b = (const float*)d_in[19];

    // workspace layout (bytes)
    char* ws = (char*)d_ws;
    unsigned short* feat1  = (unsigned short*)(ws);              // 26,214,400
    unsigned short* feat2  = (unsigned short*)(ws + 26214400);   //  3,276,800
    unsigned short* feat3  = (unsigned short*)(ws + 29491200);   //  1,048,576
    unsigned short* fout   = (unsigned short*)(ws + 30539776);   //  1,048,576
    float*          gx     = (float*)(ws + 31588352);            //  1,572,864
    unsigned short* hidb   = (unsigned short*)(ws + 33161216);   //    262,144
    unsigned short* w1b    = (unsigned short*)(ws + 33423360);   //     16,384
    unsigned short* w2b    = (unsigned short*)(ws + 33439744);   //    262,144
    unsigned short* w3b    = (unsigned short*)(ws + 33701888);   //    147,456
    unsigned short* fc1b   = (unsigned short*)(ws + 33849344);   //    524,288
    unsigned short* wcat   = (unsigned short*)(ws + 34373632);   //    393,216
    unsigned short* ucat   = (unsigned short*)(ws + 34766848);   //     98,304
    unsigned short* w3fb   = (unsigned short*)(ws + 34865152);   //      8,192

    float* out   = (float*)d_out;          // [8,128,18]
    float* hlast = out + BS * T * ACT;     // [1,8,128]

    // weight pre-conversion (bf16)
    cvt(conv1_w, w1b,  32 * 256,    32 * 256,    stream);
    cvt(conv2_w, w2b,  64 * 2048,   64 * 2048,   stream);
    cvt(conv3_w, w3b,  128 * 576,   128 * 576,   stream);
    cvt(fc1_w,   fc1b, 512 * 512,   512 * 512,   stream);
    cvt(Wz, wcat,               128 * 512, 128 * 512, stream);
    cvt(Wr, wcat + 128 * 512,   128 * 512, 128 * 512, stream);
    cvt(Wh, wcat + 256 * 512,   128 * 512, 128 * 512, stream);
    cvt(Uz, ucat,               128 * 128, 128 * 128, stream);
    cvt(Ur, ucat + 128 * 128,   128 * 128, 128 * 128, stream);
    cvt(Uh, ucat + 256 * 128,   128 * 128, 128 * 128, stream);
    cvt(fc3_w, w3fb, ACT * 128, 32 * 128, stream);

    conv1_kernel<<<dim3(NFRAMES, 25), 64,  0, stream>>>(x, w1b, conv1_b, feat1);
    conv2_kernel<<<dim3(NFRAMES),     256, 0, stream>>>(feat1, w2b, conv2_b, feat2);
    conv3_kernel<<<dim3(NFRAMES),     256, 0, stream>>>(feat2, w3b, conv3_b, feat3);
    fc1_kernel  <<<dim3(64, 8),       128, 0, stream>>>(feat3, fc1b, fc1_b, fout);
    grux_kernel <<<dim3(64, 6),       128, 0, stream>>>(fout, wcat, bz, br, bh, gx);
    gru_scan_kernel<<<dim3(1),        256, 0, stream>>>(gx, ucat, hidb, hlast);
    fc3_kernel  <<<dim3(64),          64,  0, stream>>>(hidb, w3fb, fc3_b, out);
}
